// CDRAlignedConditioner_85237920957147
// MI455X (gfx1250) — matrix operand order repaired and checked by reading, not compiled, not run on hardware
//
#include <hip/hip_runtime.h>
#include <hip/hip_bf16.h>

typedef __attribute__((ext_vector_type(16))) __bf16 v16bf;
typedef __attribute__((ext_vector_type(8)))  float  v8f;
typedef unsigned short u16;

// GCC-style vector matching the async-LDS builtin's pointee type:
//   '__attribute__((__vector_size__(4 * sizeof(int)))) int'
typedef int v4i_b128 __attribute__((vector_size(16)));

// ---------------------------------------------------------------------------
// CDNA5 async global->LDS path (guarded: falls back to sync staging if the
// toolchain doesn't declare the builtins)
// ---------------------------------------------------------------------------
#if defined(__has_builtin)
# if __has_builtin(__builtin_amdgcn_global_load_async_to_lds_b128) && \
     __has_builtin(__builtin_amdgcn_s_wait_asynccnt)
#  define USE_ASYNC_LDS 1
# endif
#endif
#ifndef USE_ASYNC_LDS
# define USE_ASYNC_LDS 0
#endif

#if USE_ASYNC_LDS
__device__ __forceinline__ void async_copy_b128(const u16* gsrc, u16* ldst) {
    // emits GLOBAL_LOAD_ASYNC_TO_LDS_B128 (ASYNCcnt-tracked DMA, no VGPR hop)
    __builtin_amdgcn_global_load_async_to_lds_b128(
        (__attribute__((address_space(1))) v4i_b128*)gsrc,
        (__attribute__((address_space(3))) v4i_b128*)ldst,
        /*imm offset*/0, /*cpol*/0);
}
#endif

// ---------------------------------------------------------------------------
// Problem constants
// ---------------------------------------------------------------------------
constexpr int BATCH  = 8;
constexpr int LTEXT  = 2048;
constexpr int NTOK   = 2048;
constexpr int CTEXT  = 3584;
constexpr int COUT   = 384;
constexpr int CHID   = 768;
constexpr int MROWS  = BATCH * LTEXT;   // 16384

// ---------------------------------------------------------------------------
// f32 -> bf16 round-to-nearest-even
// ---------------------------------------------------------------------------
__device__ __forceinline__ u16 f2bf(float f) {
    unsigned int u = __builtin_bit_cast(unsigned int, f);
    u += 0x7FFFu + ((u >> 16) & 1u);
    return (u16)(u >> 16);
}

union FragBF {            // 16 bf16 = 32 bytes = two uint4 LDS/VMEM transactions
    uint4 q[2];
    v16bf v;
};

// ---------------------------------------------------------------------------
// Convert f32 weights -> bf16 workspace
// ---------------------------------------------------------------------------
__global__ void cvt_f32_to_bf16(const float* __restrict__ src,
                                u16* __restrict__ dst, int n) {
    int i = blockIdx.x * blockDim.x + threadIdx.x;
    if (i < n) dst[i] = f2bf(src[i]);
}

// ---------------------------------------------------------------------------
// Zero-init output (float4 granularity, exact grid)
// ---------------------------------------------------------------------------
__global__ void zero_f32x4(float4* __restrict__ p) {
    size_t i = (size_t)blockIdx.x * blockDim.x + threadIdx.x;
    p[i] = make_float4(0.f, 0.f, 0.f, 0.f);
}

// ---------------------------------------------------------------------------
// bf16 WMMA GEMM:  Out[M,N] = act( A[M,K] * B[N,K]^T + bias[N] )
//   A_IS_F32:  A is f32 in global (converted to bf16 on the fly into LDS);
//              otherwise A is bf16 (u16) row-major and staged via async DMA.
//   SILU_BF16: epilogue = SiLU, output bf16; otherwise plain bias, output f32.
// Block = 256 threads (8 waves). Block tile 128x128, K step 32.
// Wave w: wm = w&3 (M strip of 32), wn = w>>2 (N strip of 64)
//   -> 2x4 grid of 16x16 accumulators per wave (64 VGPRs of f32 acc).
// ---------------------------------------------------------------------------
template <bool A_IS_F32, bool SILU_BF16>
__global__ __launch_bounds__(256)
void wmma_gemm_kernel(const void* __restrict__ Aptr,
                      const u16* __restrict__ Bw,      // [N][K] bf16
                      const float* __restrict__ bias,  // [N]
                      void* __restrict__ Out,
                      int Mdim, int Ndim, int Kdim) {
    constexpr int KB   = 32;
    constexpr int LDK  = 40;   // 32 + 8 bf16 pad (80B row stride, 16B aligned)

    __shared__ u16 As[128][LDK];
    __shared__ u16 Bs[128][LDK];

    const int tid   = threadIdx.x;
    const int lane  = tid & 31;
    const int wave  = tid >> 5;
    const int wm    = wave & 3;        // 0..3  -> M offset wm*32
    const int wn    = wave >> 2;       // 0..1  -> N offset wn*64
    const int lrow  = lane & 15;
    const int lhalf = lane >> 4;

    const int n0 = blockIdx.x * 128;
    const int m0 = blockIdx.y * 128;

    // staging assignment: each thread stages one half-row (16 elems) of A and B
    const int srow = tid >> 1;
    const int skc  = (tid & 1) << 4;   // 0 or 16

    v8f acc[2][4];
    const v8f vzero = {0.f, 0.f, 0.f, 0.f, 0.f, 0.f, 0.f, 0.f};
#pragma unroll
    for (int ms = 0; ms < 2; ++ms)
#pragma unroll
        for (int ns = 0; ns < 4; ++ns) acc[ms][ns] = vzero;

    for (int k0 = 0; k0 < Kdim; k0 += KB) {
        __syncthreads();

        // ---- stage B tile first (weights, [n][k] = K-contiguous) so the
        //      async DMA overlaps the A-tile conversion work ----
        {
            const u16* bp = Bw + (size_t)(n0 + srow) * Kdim + k0 + skc;
#if USE_ASYNC_LDS
            async_copy_b128(bp,     &Bs[srow][skc]);
            async_copy_b128(bp + 8, &Bs[srow][skc + 8]);
#else
            ((uint4*)&Bs[srow][skc])[0] = ((const uint4*)bp)[0];
            ((uint4*)&Bs[srow][skc])[1] = ((const uint4*)bp)[1];
#endif
        }
        // ---- stage A tile (128 x 32) ----
        if (A_IS_F32) {
            const float* ap = (const float*)Aptr +
                              (size_t)(m0 + srow) * Kdim + k0 + skc;
            float4 f0 = ((const float4*)ap)[0];
            float4 f1 = ((const float4*)ap)[1];
            float4 f2 = ((const float4*)ap)[2];
            float4 f3 = ((const float4*)ap)[3];
            // prefetch next K-tile of this row (emits global_prefetch_b8)
            if (k0 + KB < Kdim) __builtin_prefetch(ap + KB, 0, 1);
            alignas(16) u16 t[16] = {
                f2bf(f0.x), f2bf(f0.y), f2bf(f0.z), f2bf(f0.w),
                f2bf(f1.x), f2bf(f1.y), f2bf(f1.z), f2bf(f1.w),
                f2bf(f2.x), f2bf(f2.y), f2bf(f2.z), f2bf(f2.w),
                f2bf(f3.x), f2bf(f3.y), f2bf(f3.z), f2bf(f3.w)};
            ((uint4*)&As[srow][skc])[0] = ((const uint4*)t)[0];
            ((uint4*)&As[srow][skc])[1] = ((const uint4*)t)[1];
        } else {
            const u16* ap = (const u16*)Aptr +
                            (size_t)(m0 + srow) * Kdim + k0 + skc;
#if USE_ASYNC_LDS
            async_copy_b128(ap,     &As[srow][skc]);
            async_copy_b128(ap + 8, &As[srow][skc + 8]);
#else
            ((uint4*)&As[srow][skc])[0] = ((const uint4*)ap)[0];
            ((uint4*)&As[srow][skc])[1] = ((const uint4*)ap)[1];
#endif
        }
#if USE_ASYNC_LDS
        __builtin_amdgcn_s_wait_asynccnt(0);   // own DMA writes landed in LDS
#endif
        __syncthreads();

        // ---- fragments (ISA VGPR layouts, wave32) ----
        // A 16x32 bf16: lanes 0-15 (M=lrow) K{0-7,16-23}; lanes 16-31 K{8-15,24-31}
        FragBF afr[2];
        const int akb = lhalf * 8;
#pragma unroll
        for (int ms = 0; ms < 2; ++ms) {
            const int r = wm * 32 + ms * 16 + lrow;
            afr[ms].q[0] = *(const uint4*)&As[r][akb];
            afr[ms].q[1] = *(const uint4*)&As[r][akb + 16];
        }
        // B 32x16 bf16: lanes 0-15 (N=lrow) K{0-7,16-23}; lanes 16-31 K{8-15,24-31}
        FragBF bfr[4];
        const int bkb = lhalf * 8;
#pragma unroll
        for (int ns = 0; ns < 4; ++ns) {
            const int c = wn * 64 + ns * 16 + lrow;
            bfr[ns].q[0] = *(const uint4*)&Bs[c][bkb];
            bfr[ns].q[1] = *(const uint4*)&Bs[c][bkb + 16];
        }
        // ---- 8 WMMAs per wave per K-step ----
#pragma unroll
        for (int ms = 0; ms < 2; ++ms)
#pragma unroll
            for (int ns = 0; ns < 4; ++ns)
                acc[ms][ns] = __builtin_amdgcn_wmma_f32_16x16x32_bf16(
                    false, afr[ms].v, false, bfr[ns].v,
                    (short)0, acc[ms][ns], false, false);
    }

    // ---- epilogue: C/D layout: VGPR i -> M = i + lhalf*8, N = lrow ----
#pragma unroll
    for (int ms = 0; ms < 2; ++ms)
#pragma unroll
        for (int ns = 0; ns < 4; ++ns) {
            const int nn = n0 + wn * 64 + ns * 16 + lrow;
            const float bv = bias[nn];
#pragma unroll
            for (int i = 0; i < 8; ++i) {
                const int mm = m0 + wm * 32 + ms * 16 + lhalf * 8 + i;
                float v = acc[ms][ns][i] + bv;
                if (SILU_BF16) {
                    v = v * (1.0f / (1.0f + __expf(-v)));   // SiLU
                    ((u16*)Out)[(size_t)mm * Ndim + nn] = f2bf(v);
                } else {
                    ((float*)Out)[(size_t)mm * Ndim + nn] = v;
                }
            }
        }
}

// ---------------------------------------------------------------------------
// Rank-based CDR alignment: one workgroup per (batch, chain/CDR pair).
// Block-wide scans over the 2048 text / protein masks, then cooperative
// float4 row gather  out[b, j, :] = scale * proj[b, src, :].
// ---------------------------------------------------------------------------
__global__ __launch_bounds__(256)
void align_kernel(const float* __restrict__ proj,       // [B][L][COUT]
                  const int* __restrict__ text_mask,    // [B][L]
                  const int* __restrict__ chain_ids,    // [B][L]
                  const int* __restrict__ cdr_ids,      // [B][L]
                  const int* __restrict__ p_chain,      // [B][NTOK]
                  const int* __restrict__ p_region,     // [B][NTOK]
                  const float* __restrict__ scale,      // [1]
                  float* __restrict__ out) {            // [B][NTOK][COUT]
    const int b    = blockIdx.x / 6;
    const int pair = blockIdx.x % 6;
    const int ct   = (pair < 3) ? 1 : 2;          // chain types {1,2}
    const int rt   = 2 + 2 * (pair % 3);          // CDR types {2,4,6}
    const int t    = threadIdx.x;

    __shared__ int          sums[256];
    __shared__ u16          idx_by_rank[LTEXT];
    __shared__ unsigned int rowlist[NTOK];        // (protein_j << 16) | text_src

    // ---------- phase 1: scan text mask, build rank -> text index ----------
    const int* cB = chain_ids + (size_t)b * LTEXT;
    const int* rB = cdr_ids   + (size_t)b * LTEXT;
    const int* mB = text_mask + (size_t)b * LTEXT;
    int bits = 0, lsum = 0;
#pragma unroll
    for (int e = 0; e < 8; ++e) {
        const int i = t * 8 + e;
        const int m = (cB[i] == ct) & (rB[i] == rt) & (mB[i] != 0);
        bits |= m << e;
        lsum += m;
    }
    sums[t] = lsum;
    __syncthreads();
    for (int off = 1; off < 256; off <<= 1) {
        const int add = (t >= off) ? sums[t - off] : 0;
        __syncthreads();
        sums[t] += add;
        __syncthreads();
    }
    const int n_text = sums[255];
    int rank = sums[t] - lsum;
#pragma unroll
    for (int e = 0; e < 8; ++e)
        if ((bits >> e) & 1) idx_by_rank[rank++] = (u16)(t * 8 + e);
    __syncthreads();

    // ---------- phase 2: scan protein mask, build row copy list ----------
    const int* pcB = p_chain  + (size_t)b * NTOK;
    const int* prB = p_region + (size_t)b * NTOK;
    bits = 0; lsum = 0;
#pragma unroll
    for (int e = 0; e < 8; ++e) {
        const int j = t * 8 + e;
        const int m = (pcB[j] == ct) & (prB[j] == rt);
        bits |= m << e;
        lsum += m;
    }
    sums[t] = lsum;
    __syncthreads();
    for (int off = 1; off < 256; off <<= 1) {
        const int add = (t >= off) ? sums[t - off] : 0;
        __syncthreads();
        sums[t] += add;
        __syncthreads();
    }
    const int n_prot = sums[255];
    int pr = sums[t] - lsum;
#pragma unroll
    for (int e = 0; e < 8; ++e)
        if ((bits >> e) & 1) {
            if (pr < n_text)
                rowlist[pr] = ((unsigned)(t * 8 + e) << 16) | idx_by_rank[pr];
            ++pr;
        }
    __syncthreads();

    // ---------- phase 3: cooperative row gather ----------
    const int nrows = (n_prot < n_text) ? n_prot : n_text;
    const float s = scale[0];
    const float4* pB = (const float4*)(proj + (size_t)b * LTEXT * COUT);
    float4*       oB = (float4*)(out + (size_t)b * NTOK * COUT);
    constexpr int C4 = COUT / 4;   // 96 float4 per row
    const int total = nrows * C4;
    for (int idx = t; idx < total; idx += 256) {
        const int r = idx / C4, c = idx % C4;
        const unsigned e = rowlist[r];
        const int j = (int)(e >> 16), src = (int)(e & 0xFFFF);
        const float4 v = pB[(size_t)src * C4 + c];
        oB[(size_t)j * C4 + c] = make_float4(v.x * s, v.y * s, v.z * s, v.w * s);
    }
}

// ---------------------------------------------------------------------------
// Launch
// ---------------------------------------------------------------------------
extern "C" void kernel_launch(void* const* d_in, const int* in_sizes, int n_in,
                              void* d_out, int out_size, void* d_ws, size_t ws_size,
                              hipStream_t stream) {
    const float* text  = (const float*)d_in[0];
    const int*   tmask = (const int*)d_in[1];
    const int*   chain = (const int*)d_in[2];
    const int*   cdr   = (const int*)d_in[3];
    const int*   pch   = (const int*)d_in[4];
    const int*   prg   = (const int*)d_in[5];
    const float* W1    = (const float*)d_in[6];
    const float* b1    = (const float*)d_in[7];
    const float* W2    = (const float*)d_in[8];
    const float* b2    = (const float*)d_in[9];
    const float* scale = (const float*)d_in[10];
    float* out = (float*)d_out;

    // workspace carve-up (all 256B aligned): ~54 MB total
    char* ws = (char*)d_ws;
    u16*   W1bf = (u16*)ws;                                      // 768*3584*2
    u16*   W2bf = (u16*)(ws + (size_t)CHID * CTEXT * 2);         // 384*768*2
    u16*   hbuf = (u16*)(ws + (size_t)CHID * CTEXT * 2
                             + (size_t)COUT * CHID  * 2);        // 16384*768*2
    float* proj = (float*)(ws + (size_t)CHID * CTEXT * 2
                              + (size_t)COUT * CHID  * 2
                              + (size_t)MROWS * CHID * 2);       // 16384*384*4

    // 1) weights -> bf16
    {
        const int n1 = CHID * CTEXT;
        cvt_f32_to_bf16<<<(n1 + 255) / 256, 256, 0, stream>>>(W1, W1bf, n1);
        const int n2 = COUT * CHID;
        cvt_f32_to_bf16<<<(n2 + 255) / 256, 256, 0, stream>>>(W2, W2bf, n2);
    }
    // 2) GEMM1: h = silu(text * W1^T + b1)   [16384 x 768] bf16
    {
        dim3 grid(CHID / 128, MROWS / 128);   // (6, 128)
        wmma_gemm_kernel<true, true><<<grid, 256, 0, stream>>>(
            (const void*)text, W1bf, b1, (void*)hbuf, MROWS, CHID, CTEXT);
    }
    // 3) GEMM2: proj = h * W2^T + b2         [16384 x 384] f32
    {
        dim3 grid(COUT / 128, MROWS / 128);   // (3, 128)
        wmma_gemm_kernel<false, false><<<grid, 256, 0, stream>>>(
            (const void*)hbuf, W2bf, b2, (void*)proj, MROWS, COUT, CHID);
    }
    // 4) zero output, then rank-aligned gather
    {
        const size_t n4 = (size_t)BATCH * NTOK * COUT / 4;       // 1572864
        zero_f32x4<<<(unsigned)(n4 / 256), 256, 0, stream>>>((float4*)out);
        align_kernel<<<BATCH * 6, 256, 0, stream>>>(
            proj, tmask, chain, cdr, pch, prg, scale, out);
    }
}
